// selfattention_4569845203261
// MI455X (gfx1250) — matrix operand order repaired and checked by reading, not compiled, not run on hardware
//
#include <hip/hip_runtime.h>

// ---- CDNA5 WMMA types ----
typedef __attribute__((ext_vector_type(16))) _Float16 v16h;
typedef __attribute__((ext_vector_type(8)))  float    v8f;
typedef int v4i __attribute__((vector_size(16)));

#define AS1 __attribute__((address_space(1)))
#define AS3 __attribute__((address_space(3)))

#define B_ 2
#define H_ 16
#define L_ 2048
#define D_ 128

#define BM 64            // query rows per block
#define KT 32            // key rows per j-step
#define JT (L_ / KT)     // 64 j-steps

#define QS_STRIDE 132    // Q tile row stride (f16): dword stores, bank-safe
#define KS_STRIDE 136    // K tile row stride: 272B -> 16B aligned rows, bank-safe
#define VT_STRIDE 40     // V^T tile row stride: 80B -> 16B aligned rows, bank-safe
#define SS_STRIDE 36     // S tile row stride

#if defined(__AMDGCN__) && __has_builtin(__builtin_amdgcn_global_load_async_to_lds_b128) && __has_builtin(__builtin_amdgcn_s_wait_asynccnt)
#define HAVE_ASYNC 1
#else
#define HAVE_ASYNC 0
#endif

union Frag {
  unsigned int u[8];
  v16h v;
};

// two f32 -> packed f16 in ONE VALU op (v_cvt_pk_rtz_f16_f32)
__device__ __forceinline__ unsigned int pack2h(float lo, float hi) {
  return __builtin_bit_cast(unsigned int, __builtin_amdgcn_cvt_pkrtz(lo, hi));
}

// A-operand (16x32 f16) per ISA layout: VGPR v holds K = 2v + (v>=4?8:0) + 8*half
__device__ __forceinline__ void load_fragA(const unsigned short* rowbase, int half, Frag& f) {
#pragma unroll
  for (int v = 0; v < 8; ++v) {
    int k = 2 * v + (v >= 4 ? 8 : 0) + 8 * half;
    f.u[v] = *(const unsigned int*)(rowbase + k);
  }
}

// B-operand (32x16 f16): same K order as A: VGPR v holds K = 2v + (v>=4?8:0) + 8*half
__device__ __forceinline__ void load_fragB(const unsigned short* rowbase, int half, Frag& f) {
#pragma unroll
  for (int v = 0; v < 8; ++v) {
    f.u[v] = *(const unsigned int*)(rowbase + 2 * v + (v >= 4 ? 8 : 0) + 8 * half);
  }
}

// ===========================================================================
// Prep kernel: K (f32, [bh][L][D]) -> khf (f16, same layout)
//              V (f32, [bh][L][D]) -> vtb (f16, TRANSPOSED [bh][D][L])
// Done once; amortizes conversion over the 32 blocks that share each bh.
// ===========================================================================
__global__ __launch_bounds__(256)
void prep_kv_f16(const float* __restrict__ k, const float* __restrict__ v,
                 unsigned short* __restrict__ khf, unsigned short* __restrict__ vtb) {
  __shared__ __align__(16) unsigned short VtL[D_ * VT_STRIDE];

  const int tid = threadIdx.x;
  const int jt  = blockIdx.x;  // 0..63 key tiles
  const int bh  = blockIdx.y;
  const size_t base = ((size_t)bh * L_ + (size_t)jt * KT) * D_;

  const float4* kg = (const float4*)(k + base);
  const float4* vg = (const float4*)(v + base);
  uint2* ko = (uint2*)(khf + base);

#pragma unroll
  for (int i = 0; i < 4; ++i) {
    int idx = tid + 256 * i;     // 0..1023 float4 of the 32x128 tile
    float4 xk = kg[idx];
    uint2 p; p.x = pack2h(xk.x, xk.y); p.y = pack2h(xk.z, xk.w);
    ko[idx] = p;

    int row = idx >> 5;          // 0..31
    int c4  = idx & 31;
    float4 xv = vg[idx];
    unsigned int p0 = pack2h(xv.x, xv.y);
    unsigned int p1 = pack2h(xv.z, xv.w);
    VtL[(c4 * 4 + 0) * VT_STRIDE + row] = (unsigned short)(p0 & 0xFFFFu);
    VtL[(c4 * 4 + 1) * VT_STRIDE + row] = (unsigned short)(p0 >> 16);
    VtL[(c4 * 4 + 2) * VT_STRIDE + row] = (unsigned short)(p1 & 0xFFFFu);
    VtL[(c4 * 4 + 3) * VT_STRIDE + row] = (unsigned short)(p1 >> 16);
  }
  __syncthreads();

  // write transposed tile: row d has 32 contiguous j's (64B)
  int r = tid >> 1, h = tid & 1;
  const uint4* src = (const uint4*)&VtL[r * VT_STRIDE + h * 16];
  uint4 a = src[0], b = src[1];
  uint4* dst = (uint4*)(vtb + ((size_t)bh * D_ + r) * L_ + (size_t)jt * KT + h * 16);
  dst[0] = a; dst[1] = b;
}

// ---- stage one K/V tile (f16) into LDS: async-to-LDS if available ----
__device__ __forceinline__ void stage_tile(const unsigned short* kg, const unsigned short* vgT,
                                           unsigned short* ksDst, unsigned short* vtDst,
                                           int tid, int jt) {
  // K tile: 32x128 f16 = 512 b128 chunks (16B), 2 per thread
#pragma unroll
  for (int i = 0; i < 2; ++i) {
    int c = tid * 2 + i;
    int row = c >> 4, e = (c & 15) * 8;
    const unsigned short* g = kg + (size_t)(jt * KT + row) * D_ + e;
    unsigned short* l = ksDst + row * KS_STRIDE + e;
#if HAVE_ASYNC
    __builtin_amdgcn_global_load_async_to_lds_b128((AS1 v4i*)g, (AS3 v4i*)l, 0, 0);
#else
    *(uint4*)l = *(const uint4*)g;
#endif
  }
  // V^T tile: 128x32 f16 = 512 b128 chunks, 2 per thread
#pragma unroll
  for (int i = 0; i < 2; ++i) {
    int c = tid * 2 + i;
    int row = c >> 2, e = (c & 3) * 8;
    const unsigned short* g = vgT + (size_t)row * L_ + jt * KT + e;
    unsigned short* l = vtDst + row * VT_STRIDE + e;
#if HAVE_ASYNC
    __builtin_amdgcn_global_load_async_to_lds_b128((AS1 v4i*)g, (AS3 v4i*)l, 0, 0);
#else
    *(uint4*)l = *(const uint4*)g;
#endif
  }
}

// ===========================================================================
// Main kernel: double-buffered async pipeline + WMMA
// ===========================================================================
__global__ __launch_bounds__(256)
void attn_main_wmma(const float* __restrict__ q,
                    const unsigned short* __restrict__ khf,
                    const unsigned short* __restrict__ vtb,
                    float* __restrict__ out) {
  __shared__ __align__(16) unsigned short Qs[BM * QS_STRIDE];
  __shared__ __align__(16) unsigned short Ks[2][KT * KS_STRIDE];
  __shared__ __align__(16) unsigned short Vt[2][D_ * VT_STRIDE];
  __shared__ __align__(16) unsigned short Ss[BM * SS_STRIDE];

  const int tid  = threadIdx.x;
  const int wave = tid >> 5;
  const int lane = tid & 31;
  const int half = lane >> 4;
  const int lm   = lane & 15;
  const int mi   = wave & 3;
  const int nt   = wave >> 2;

  const int m0 = blockIdx.x * BM;
  const int bh = blockIdx.y;
  const size_t base = (size_t)bh * L_ * D_;
  const float scale = 0.08838834764831845f;  // 1/sqrt(128), folded into Q

  // ---- Q tile -> LDS (f16, scaled) ----
  {
    const float4* qg = (const float4*)(q + base + (size_t)m0 * D_);
#pragma unroll
    for (int i = 0; i < 8; ++i) {
      int idx = tid + 256 * i;
      int row = idx >> 5, c4 = idx & 31;
      float4 x = qg[idx];
      unsigned int* dst = (unsigned int*)&Qs[row * QS_STRIDE + c4 * 4];
      dst[0] = pack2h(x.x * scale, x.y * scale);
      dst[1] = pack2h(x.z * scale, x.w * scale);
    }
  }

  const unsigned short* kg  = khf + base;
  const unsigned short* vgT = vtb + base;  // [D][L] for this bh

  stage_tile(kg, vgT, Ks[0], Vt[0], tid, 0);   // prologue: tile 0 in flight

  __syncthreads();  // Qs visible

  Frag qa[4];
#pragma unroll
  for (int c = 0; c < 4; ++c)
    load_fragA(&Qs[(mi * 16 + lm) * QS_STRIDE + c * 32], half, qa[c]);

  v8f oacc[4];
#pragma unroll
  for (int t = 0; t < 4; ++t) oacc[t] = (v8f){};

  for (int jt = 0; jt < JT; ++jt) {
    const int cur = jt & 1;
    // issue next tile's copies (into buffer consumed in the previous iteration)
    if (jt + 1 < JT) {
      stage_tile(kg, vgT, Ks[cur ^ 1], Vt[cur ^ 1], tid, jt + 1);
#if HAVE_ASYNC
      __builtin_amdgcn_s_wait_asynccnt(4);   // tile jt landed; jt+1 still in flight
#endif
    } else {
#if HAVE_ASYNC
      __builtin_amdgcn_s_wait_asynccnt(0);
#endif
    }
    __syncthreads();   // all waves' tile-jt data visible

    // ---- S = (Q*scale) @ K^T : one 16x16 score tile per wave, K-dim=128 ----
    v8f sacc = (v8f){};
#pragma unroll
    for (int c = 0; c < 4; ++c) {
      Frag kb;
      load_fragB(&Ks[cur][(nt * 16 + lm) * KS_STRIDE + c * 32], half, kb);
      sacc = __builtin_amdgcn_wmma_f32_16x16x32_f16(false, qa[c].v, false, kb.v,
                                                    (short)0, sacc, false, false);
    }

    // ---- spill S tile to LDS as f16 (C layout -> row-major) ----
#pragma unroll
    for (int i = 0; i < 4; ++i) {
      unsigned int p = pack2h(sacc[2 * i], sacc[2 * i + 1]);
      int row = mi * 16 + 2 * i + 8 * half;       // pair -> rows row, row+1
      Ss[row * SS_STRIDE + nt * 16 + lm]       = (unsigned short)(p & 0xFFFFu);
      Ss[(row + 1) * SS_STRIDE + nt * 16 + lm] = (unsigned short)(p >> 16);
    }
    __syncthreads();

    // ---- O += S @ V : four 16x16 output tiles per wave, K-dim=32 ----
    Frag sa;
    load_fragA(&Ss[(mi * 16 + lm) * SS_STRIDE], half, sa);
#pragma unroll
    for (int t = 0; t < 4; ++t) {
      Frag vb;
      load_fragB(&Vt[cur][(nt * 64 + t * 16 + lm) * VT_STRIDE], half, vb);
      oacc[t] = __builtin_amdgcn_wmma_f32_16x16x32_f16(false, sa.v, false, vb.v,
                                                       (short)0, oacc[t], false, false);
    }
    __syncthreads();   // buffer 'cur' + Ss free for reuse
  }

  // ---- write O (f32, C layout) ----
  float* og = out + base + (size_t)(m0 + mi * 16) * D_;
#pragma unroll
  for (int t = 0; t < 4; ++t) {
#pragma unroll
    for (int r = 0; r < 8; ++r) {
      og[(size_t)(r + 8 * half) * D_ + nt * 64 + t * 16 + lm] = oacc[t][r];
    }
  }
}

// ===========================================================================
// Fallback (small workspace): self-contained kernel (f32 -> f16 on the fly)
// ===========================================================================
__global__ __launch_bounds__(256)
void attn_mono_wmma(const float* __restrict__ q,
                    const float* __restrict__ k,
                    const float* __restrict__ v,
                    float* __restrict__ out) {
  __shared__ unsigned short Qs[BM * QS_STRIDE];
  __shared__ unsigned short KsM[KT * QS_STRIDE];
  __shared__ unsigned short VtM[D_ * SS_STRIDE];
  __shared__ unsigned short SsM[BM * SS_STRIDE];

  const int tid  = threadIdx.x;
  const int wave = tid >> 5;
  const int lane = tid & 31;
  const int half = lane >> 4;
  const int lm   = lane & 15;
  const int mi   = wave & 3;
  const int nt   = wave >> 2;

  const int m0 = blockIdx.x * BM;
  const int bh = blockIdx.y;
  const size_t base = (size_t)bh * L_ * D_;
  const float scale = 0.08838834764831845f;

  {
    const float4* qg = (const float4*)(q + base + (size_t)m0 * D_);
#pragma unroll
    for (int i = 0; i < 8; ++i) {
      int idx = tid + 256 * i;
      int row = idx >> 5, c4 = idx & 31;
      float4 x = qg[idx];
      unsigned int* dst = (unsigned int*)&Qs[row * QS_STRIDE + c4 * 4];
      dst[0] = pack2h(x.x * scale, x.y * scale);
      dst[1] = pack2h(x.z * scale, x.w * scale);
    }
  }
  __syncthreads();

  Frag qa[4];
#pragma unroll
  for (int c = 0; c < 4; ++c)
    load_fragA(&Qs[(mi * 16 + lm) * QS_STRIDE + c * 32], half, qa[c]);

  v8f oacc[4];
#pragma unroll
  for (int t = 0; t < 4; ++t) oacc[t] = (v8f){};

  const float4* kgm = (const float4*)(k + base);
  const float4* vgm = (const float4*)(v + base);

  for (int jt = 0; jt < JT; ++jt) {
    __syncthreads();
#pragma unroll
    for (int i = 0; i < 4; ++i) {
      int idx = tid + 256 * i;
      int row = idx >> 5, c4 = idx & 31;
      float4 xk = kgm[(size_t)(jt * KT) * 32 + idx];
      unsigned int* dk = (unsigned int*)&KsM[row * QS_STRIDE + c4 * 4];
      dk[0] = pack2h(xk.x, xk.y);
      dk[1] = pack2h(xk.z, xk.w);
      float4 xv = vgm[(size_t)(jt * KT) * 32 + idx];
      unsigned int p0 = pack2h(xv.x, xv.y);
      unsigned int p1 = pack2h(xv.z, xv.w);
      VtM[(c4 * 4 + 0) * SS_STRIDE + row] = (unsigned short)(p0 & 0xFFFFu);
      VtM[(c4 * 4 + 1) * SS_STRIDE + row] = (unsigned short)(p0 >> 16);
      VtM[(c4 * 4 + 2) * SS_STRIDE + row] = (unsigned short)(p1 & 0xFFFFu);
      VtM[(c4 * 4 + 3) * SS_STRIDE + row] = (unsigned short)(p1 >> 16);
    }
    __syncthreads();

    v8f sacc = (v8f){};
#pragma unroll
    for (int c = 0; c < 4; ++c) {
      Frag kb;
      load_fragB(&KsM[(nt * 16 + lm) * QS_STRIDE + c * 32], half, kb);
      sacc = __builtin_amdgcn_wmma_f32_16x16x32_f16(false, qa[c].v, false, kb.v,
                                                    (short)0, sacc, false, false);
    }
#pragma unroll
    for (int i = 0; i < 4; ++i) {
      unsigned int p = pack2h(sacc[2 * i], sacc[2 * i + 1]);
      int row = mi * 16 + 2 * i + 8 * half;
      SsM[row * SS_STRIDE + nt * 16 + lm]       = (unsigned short)(p & 0xFFFFu);
      SsM[(row + 1) * SS_STRIDE + nt * 16 + lm] = (unsigned short)(p >> 16);
    }
    __syncthreads();

    Frag sa;
    load_fragA(&SsM[(mi * 16 + lm) * SS_STRIDE], half, sa);
#pragma unroll
    for (int t = 0; t < 4; ++t) {
      Frag vb;
      load_fragB(&VtM[(nt * 64 + t * 16 + lm) * SS_STRIDE], half, vb);
      oacc[t] = __builtin_amdgcn_wmma_f32_16x16x32_f16(false, sa.v, false, vb.v,
                                                       (short)0, oacc[t], false, false);
    }
  }

  float* og = out + base + (size_t)(m0 + mi * 16) * D_;
#pragma unroll
  for (int t = 0; t < 4; ++t)
#pragma unroll
    for (int r = 0; r < 8; ++r)
      og[(size_t)(r + 8 * half) * D_ + nt * 64 + t * 16 + lm] = oacc[t][r];
}

extern "C" void kernel_launch(void* const* d_in, const int* in_sizes, int n_in,
                              void* d_out, int out_size, void* d_ws, size_t ws_size,
                              hipStream_t stream) {
  (void)in_sizes; (void)n_in; (void)out_size;
  const float* q = (const float*)d_in[0];
  const float* k = (const float*)d_in[1];
  const float* v = (const float*)d_in[2];
  float* out = (float*)d_out;

  const size_t elems = (size_t)B_ * H_ * L_ * D_;
  const size_t need  = 2 * elems * sizeof(unsigned short);  // khf + vtb

  if (d_ws != nullptr && ws_size >= need) {
    unsigned short* khf = (unsigned short*)d_ws;
    unsigned short* vtb = khf + elems;
    hipLaunchKernelGGL(prep_kv_f16, dim3(JT, B_ * H_), dim3(256), 0, stream, k, v, khf, vtb);
    hipLaunchKernelGGL(attn_main_wmma, dim3(L_ / BM, B_ * H_), dim3(256), 0, stream,
                       q, (const unsigned short*)khf, (const unsigned short*)vtb, out);
  } else {
    hipLaunchKernelGGL(attn_mono_wmma, dim3(L_ / BM, B_ * H_), dim3(256), 0, stream, q, k, v, out);
  }
}
